// GatheringLoss_7739531067606
// MI455X (gfx1250) — matrix operand order repaired and checked by reading, not compiled, not run on hardware
//
#include <hip/hip_runtime.h>
#include <hip/hip_bf16.h>

typedef __attribute__((ext_vector_type(16))) _Float16 v16h;
typedef __attribute__((ext_vector_type(8)))  _Float16 v8h;
typedef __attribute__((ext_vector_type(4)))  _Float16 v4h;
typedef __attribute__((ext_vector_type(8)))  float    v8f;
typedef __attribute__((ext_vector_type(4)))  float    v4f;

#define C_DIM           128   // feature dim (K of the GEMM)
#define ROWS_PER_WAVE   32    // two 16-row WMMA tiles per wave (2x B reuse)
#define ROWS_PER_BLOCK  256   // 8 waves x 32 rows
#define ITEMS_PER_CHUNK 128   // items staged per LDS chunk
#define LDS_PITCH       136   // f16 per LDS row: 128 + 8 pad (breaks bank alignment)

__global__ void gl_init_acc(double* acc) { *acc = 0.0; }

__global__ void gl_finalize(const double* acc, float* out, double count) {
    *out = (float)(*acc / count);
}

__global__ void __launch_bounds__(256)
gl_main(const float* __restrict__ q, const float* __restrict__ items,
        int M, double* __restrict__ acc)
{
    __shared__ _Float16 smem[ITEMS_PER_CHUNK * LDS_PITCH];

    const int tid  = threadIdx.x;
    const int wave = tid >> 5;
    const int lane = tid & 31;
    const int l15  = lane & 15;
    const int half = lane >> 4;

    const int rowBase = blockIdx.x * ROWS_PER_BLOCK + wave * ROWS_PER_WAVE;

    // ---- Build this wave's A fragments for TWO 16-row tiles: f32 -> f16,
    // ISA 16-bit A layout. Lane holds row (lane&15); elems 0..7 = cols
    // colBase..+7, elems 8..15 = colBase+16..+23, colBase = 32*ks + 8*half.
    v16h afrag[2][4];
    #pragma unroll
    for (int u = 0; u < 2; ++u) {
        const float* arow = q + (size_t)(rowBase + u * 16 + l15) * C_DIM;
        #pragma unroll
        for (int ks = 0; ks < 4; ++ks) {
            const int colBase = ks * 32 + half * 8;
            v4f a0 = *(const v4f*)(arow + colBase);
            v4f a1 = *(const v4f*)(arow + colBase + 4);
            v4f a2 = *(const v4f*)(arow + colBase + 16);
            v4f a3 = *(const v4f*)(arow + colBase + 20);
            v16h f;
            #pragma unroll
            for (int j = 0; j < 4; ++j) {
                f[j]      = (_Float16)a0[j];
                f[4 + j]  = (_Float16)a1[j];
                f[8 + j]  = (_Float16)a2[j];
                f[12 + j] = (_Float16)a3[j];
            }
            afrag[u][ks] = f;
        }
    }

    // Running top-1 per row, kept in the WMMA C/D layout:
    // slot [u][r] <-> query row (u*16 + r + 8*half), lane <-> candidate N = l15.
    float bestVal[2][8];
    int   bestIdx[2][8];
    #pragma unroll
    for (int u = 0; u < 2; ++u)
        #pragma unroll
        for (int r = 0; r < 8; ++r) { bestVal[u][r] = -__builtin_inff(); bestIdx[u][r] = 0; }

    const int nChunks = M / ITEMS_PER_CHUNK;
    for (int ch = 0; ch < nChunks; ++ch) {
        __syncthreads();
        // ---- Stage items chunk f32 -> f16 into LDS (coalesced float4 loads).
        const float* cbase = items + (size_t)ch * ITEMS_PER_CHUNK * C_DIM;
        #pragma unroll
        for (int i = 0; i < (ITEMS_PER_CHUNK * C_DIM / 4) / 256; ++i) {
            int e4 = i * 256 + tid;          // float4 index within chunk
            int r  = e4 >> 5;                // 32 float4 per item row
            int c4 = e4 & 31;
            v4f v = *(const v4f*)(cbase + r * C_DIM + c4 * 4);
            v4h h;
            #pragma unroll
            for (int j = 0; j < 4; ++j) h[j] = (_Float16)v[j];
            *(v4h*)(&smem[r * LDS_PITCH + c4 * 4]) = h;
        }
        __syncthreads();

        // Prefetch next chunk into cache while we compute on this one.
        if (ch + 1 < nChunks)
            __builtin_prefetch(items + (size_t)(ch + 1) * ITEMS_PER_CHUNK * C_DIM + tid * 64, 0, 0);

        // ---- 8 item tiles; per tile: load ALL 4 K-step B fragments (8x
        // ds_load_b128, one wait), then 8 WMMAs back-to-back, alternating the
        // two accumulators so each B fragment feeds 2 WMMAs with no RAW chain.
        #pragma unroll
        for (int t = 0; t < 8; ++t) {
            // B layout: lane holds item column N = l15, K = 8*half + e (e<8), 16 + 8*half + (e-8) (e>=8).
            const _Float16* brow = &smem[(t * 16 + l15) * LDS_PITCH + half * 8];
            v16h bfrag[4];
            #pragma unroll
            for (int ks = 0; ks < 4; ++ks) {
                v8h b0 = *(const v8h*)(brow + ks * 32);
                v8h b1 = *(const v8h*)(brow + ks * 32 + 16);
                bfrag[ks] = __builtin_shufflevector(b0, b1,
                            0,1,2,3,4,5,6,7,8,9,10,11,12,13,14,15);
            }
            v8f c0 = {};
            v8f c1 = {};
            #pragma unroll
            for (int ks = 0; ks < 4; ++ks) {
                c0 = __builtin_amdgcn_wmma_f32_16x16x32_f16(
                         false, afrag[0][ks], false, bfrag[ks], (short)0, c0, false, false);
                c1 = __builtin_amdgcn_wmma_f32_16x16x32_f16(
                         false, afrag[1][ks], false, bfrag[ks], (short)0, c1, false, false);
            }
            const int itemIdx = ch * ITEMS_PER_CHUNK + t * 16 + l15;
            #pragma unroll
            for (int r = 0; r < 8; ++r) {
                if (c0[r] > bestVal[0][r]) { bestVal[0][r] = c0[r]; bestIdx[0][r] = itemIdx; }
                if (c1[r] > bestVal[1][r]) { bestVal[1][r] = c1[r]; bestIdx[1][r] = itemIdx; }
            }
        }
    }

    // ---- Cross-lane argmax over the 16 lanes of each half (xor stays in-half).
    // Tie-break: lowest index (jnp.argmax picks the first maximum).
    #pragma unroll
    for (int u = 0; u < 2; ++u) {
        #pragma unroll
        for (int r = 0; r < 8; ++r) {
            float v = bestVal[u][r];
            int   ix = bestIdx[u][r];
            #pragma unroll
            for (int m = 1; m <= 8; m <<= 1) {
                float ov = __shfl_xor(v, m, 32);
                int   oi = __shfl_xor(ix, m, 32);
                if (ov > v || (ov == v && oi < ix)) { v = ov; ix = oi; }
            }
            bestIdx[u][r] = ix;
        }
    }

    // ---- MSE in full f32 from global memory. Row (u*16 + m)'s index lives in
    // slot [u][m&7] of half (m>>3); broadcast from lane 0 / lane 16.
    float sse = 0.0f;
    const int c0col = lane * 4;
    #pragma unroll
    for (int u = 0; u < 2; ++u) {
        #pragma unroll
        for (int r = 0; r < 8; ++r) {
            int idxLo = __shfl(bestIdx[u][r], 0, 32);   // row u*16 + r
            int idxHi = __shfl(bestIdx[u][r], 16, 32);  // row u*16 + r + 8
            {
                v4f qa = *(const v4f*)(q + (size_t)(rowBase + u * 16 + r) * C_DIM + c0col);
                v4f ia = *(const v4f*)(items + (size_t)idxLo * C_DIM + c0col);
                #pragma unroll
                for (int j = 0; j < 4; ++j) { float d = qa[j] - ia[j]; sse += d * d; }
            }
            {
                v4f qa = *(const v4f*)(q + (size_t)(rowBase + u * 16 + r + 8) * C_DIM + c0col);
                v4f ia = *(const v4f*)(items + (size_t)idxHi * C_DIM + c0col);
                #pragma unroll
                for (int j = 0; j < 4; ++j) { float d = qa[j] - ia[j]; sse += d * d; }
            }
        }
    }
    #pragma unroll
    for (int m = 16; m >= 1; m >>= 1) sse += __shfl_xor(sse, m, 32);
    if (lane == 0) atomicAdd(acc, (double)sse);   // global_atomic_add_f64
}

extern "C" void kernel_launch(void* const* d_in, const int* in_sizes, int n_in,
                              void* d_out, int out_size, void* d_ws, size_t ws_size,
                              hipStream_t stream) {
    (void)n_in; (void)out_size; (void)ws_size;
    const float* q     = (const float*)d_in[0];   // (64, 2048, 128) f32
    const float* items = (const float*)d_in[1];   // (1024, 128) f32
    const int Trows = in_sizes[0] / C_DIM;        // 131072
    const int M     = in_sizes[1] / C_DIM;        // 1024
    double* acc = (double*)d_ws;

    gl_init_acc<<<1, 1, 0, stream>>>(acc);
    gl_main<<<Trows / ROWS_PER_BLOCK, 256, 0, stream>>>(q, items, M, acc);
    gl_finalize<<<1, 1, 0, stream>>>(acc, (float*)d_out,
                                     (double)Trows * (double)C_DIM);
}
